// TGN_19499151524022
// MI455X (gfx1250) — matrix operand order repaired and checked by reading, not compiled, not run on hardware
//
#include <hip/hip_runtime.h>
#include <hip/hip_bf16.h>

typedef __attribute__((ext_vector_type(16))) _Float16 v16h;
typedef __attribute__((ext_vector_type(8)))  _Float16 v8h;
typedef __attribute__((ext_vector_type(4)))  _Float16 v4h;
typedef __attribute__((ext_vector_type(8)))  float    v8f;

#define DSZ   128   // D_S == D_E == D_M
#define DHID  192
#define K1    384   // layer-1 K without the delta_t column
#define KT1   12    // 384/32
#define NT1   12    // 192/16
#define KT2   6     // 192/32
#define NT2   8     // 128/16
#define KTG   4     // 128/32
#define NTG   24    // 384/16

#define TE    64    // edges per block (4 M-tiles) -> B-fragment reuse x4
#define MT    4
#define TN    32    // nodes per block (2 M-tiles) -> B-fragment reuse x2
#define MTG   2

// ---- helpers -------------------------------------------------------------

// Build a 16x32 f16 A-fragment for one lane from an LDS row.
// Lane l holds row M = l&15; K base offset c = (l>=16)?8:0.
// VGPRs 0-3 = K c..c+7 (contiguous), VGPRs 4-7 = K c+16..c+23 (contiguous).
static __device__ __forceinline__ v16h mk_a(const _Float16* p) {
  v8h lo = *(const v8h*)p;         // ds_load_b128
  v8h hi = *(const v8h*)(p + 16);  // ds_load_b128
  v16h a;
#pragma unroll
  for (int i = 0; i < 8; ++i) { a[i] = lo[i]; a[i + 8] = hi[i]; }
  return a;
}

// Pre-swizzled B fragment: 32 lanes x 16 halfs, contiguous per lane.
static __device__ __forceinline__ v16h ld_b(const _Float16* base, int frag, int lane) {
  return *(const v16h*)(base + ((size_t)frag * 32u + (size_t)lane) * 16u);
}

// ---- workspace init ------------------------------------------------------

__global__ void tgn_zero(float* __restrict__ p, size_t n) {
  size_t i = (size_t)blockIdx.x * blockDim.x + threadIdx.x;
  size_t stride = (size_t)gridDim.x * blockDim.x;
  for (; i < n; i += stride) p[i] = 0.0f;
}

// ---- weight pre-pack: f32 [OUT,IN] row-major -> f16 WMMA B fragments -----
__global__ void tgn_pack(const float* __restrict__ W, _Float16* __restrict__ dstp,
                         int KT, int NT, int stride, int skip_col) {
  int total = KT * NT * 512;
  for (int i = blockIdx.x * blockDim.x + threadIdx.x; i < total;
       i += gridDim.x * blockDim.x) {
    int j    = i & 15;
    int lane = (i >> 4) & 31;
    int frag = i >> 9;
    int nt = frag % NT;
    int kt = frag / NT;
    int K  = kt * 32 + ((lane >= 16) ? 8 : 0) + ((j < 8) ? j : (j + 8));
    int Nn = nt * 16 + (lane & 15);
    int col = (skip_col >= 0 && K >= skip_col) ? (K + 1) : K;
    dstp[i] = (_Float16)W[(size_t)Nn * (size_t)stride + (size_t)col];
  }
}

// ---- edge message kernel: 64 edges / block, 4 waves ----------------------

__launch_bounds__(128)
__global__ void tgn_edge(const float* __restrict__ S,
                         const int* __restrict__ src,
                         const int* __restrict__ dst,
                         const float* __restrict__ dt,
                         const float* __restrict__ ea,
                         const float* __restrict__ W1,   // f32, delta_t column
                         const float* __restrict__ b1,
                         const float* __restrict__ b2,
                         const _Float16* __restrict__ w1f,
                         const _Float16* __restrict__ w2f,
                         float* __restrict__ msum,
                         float* __restrict__ cnt,
                         int E) {
  __shared__ _Float16 lds_rm[TE * K1];    // 48 KB
  __shared__ _Float16 lds_h[TE * DHID];   // 24 KB
  __shared__ float    lds_dt[TE];
  __shared__ int      lds_dst[TE];

  const int tid   = threadIdx.x;
  const int lane  = tid & 31;
  const int w     = tid >> 5;
  const int ebase = blockIdx.x * TE;

  if (tid < TE) {
    int e = ebase + tid; if (e >= E) e = E - 1;
    lds_dt[tid]  = dt[e];
    lds_dst[tid] = dst[e];
  }

  // Stage three 64x128 f32 blocks as f16 into LDS (coalesced float4 loads).
#pragma unroll
  for (int p = 0; p < 3; ++p) {
#pragma unroll
    for (int i = 0; i < 16; ++i) {
      int q   = i * 128 + tid;          // 0..2047
      int row = q >> 5, c4 = q & 31;
      int e = ebase + row; if (e >= E) e = E - 1;
      const float4* sp;
      if (p == 0)      sp = (const float4*)(S  + (size_t)dst[e] * DSZ);
      else if (p == 1) sp = (const float4*)(S  + (size_t)src[e] * DSZ);
      else             sp = (const float4*)(ea + (size_t)e      * DSZ);
      float4 v = sp[c4];
      v4h h = {(_Float16)v.x, (_Float16)v.y, (_Float16)v.z, (_Float16)v.w};
      *(v4h*)&lds_rm[row * K1 + p * 128 + c4 * 4] = h;   // ds_store_b64
    }
  }
  __syncthreads();

  const int m    = lane & 15;
  const int cb   = (lane >= 16) ? 8 : 0;
  const int mrow = (lane >= 16) ? 8 : 0;

  // ---- layer 1: hmid = relu(rm @ W1.T + b1); 3 N-tiles x 4 M-tiles/wave ----
  v8f acc1[3][MT] = {};
  for (int kt = 0; kt < KT1; ++kt) {
    v16h a[MT];
#pragma unroll
    for (int mt = 0; mt < MT; ++mt)
      a[mt] = mk_a(&lds_rm[(mt * 16 + m) * K1 + kt * 32 + cb]);
#pragma unroll
    for (int j = 0; j < 3; ++j) {
      v16h b = ld_b(w1f, kt * NT1 + (w * 3 + j), lane);
#pragma unroll
      for (int mt = 0; mt < MT; ++mt)
        acc1[j][mt] = __builtin_amdgcn_wmma_f32_16x16x32_f16(
            false, a[mt], false, b, (short)0, acc1[j][mt], false, false);
    }
  }
#pragma unroll
  for (int j = 0; j < 3; ++j) {
    int nt = w * 3 + j;
    int hc = nt * 16 + (lane & 15);
    float wdt = W1[(size_t)hc * 385 + 256];
    float bb  = b1[hc];
#pragma unroll
    for (int mt = 0; mt < MT; ++mt) {
#pragma unroll
      for (int v = 0; v < 8; ++v) {
        int mm  = mt * 16 + v + mrow;          // edge row in block tile
        float val = acc1[j][mt][v] + bb + lds_dt[mm] * wdt;
        val = fmaxf(val, 0.0f);                // ReLU
        lds_h[mm * DHID + hc] = (_Float16)val;
      }
    }
  }
  __syncthreads();

  // ---- layer 2: m = hmid @ W2.T + b2; 2 N-tiles x 4 M-tiles/wave ----
  v8f acc2[2][MT] = {};
  for (int kt = 0; kt < KT2; ++kt) {
    v16h a[MT];
#pragma unroll
    for (int mt = 0; mt < MT; ++mt)
      a[mt] = mk_a(&lds_h[(mt * 16 + m) * DHID + kt * 32 + cb]);
#pragma unroll
    for (int j = 0; j < 2; ++j) {
      v16h b = ld_b(w2f, kt * NT2 + (w * 2 + j), lane);
#pragma unroll
      for (int mt = 0; mt < MT; ++mt)
        acc2[j][mt] = __builtin_amdgcn_wmma_f32_16x16x32_f16(
            false, a[mt], false, b, (short)0, acc2[j][mt], false, false);
    }
  }
  // scatter-add messages into msum
#pragma unroll
  for (int j = 0; j < 2; ++j) {
    int col = (w * 2 + j) * 16 + (lane & 15);
    float bb = b2[col];
#pragma unroll
    for (int mt = 0; mt < MT; ++mt) {
#pragma unroll
      for (int v = 0; v < 8; ++v) {
        int mm = mt * 16 + v + mrow;
        if (ebase + mm < E) {
          float val = acc2[j][mt][v] + bb;
          atomicAdd(&msum[(size_t)lds_dst[mm] * DSZ + col], val);
        }
      }
    }
  }
  if (tid < TE && ebase + tid < E) atomicAdd(&cnt[lds_dst[tid]], 1.0f);
}

// ---- GRU memory-updater kernel: 32 nodes / block, 8 waves ----------------

__launch_bounds__(256)
__global__ void tgn_gru(const float* __restrict__ S,
                        const float* __restrict__ msum,
                        const float* __restrict__ cnt,
                        const _Float16* __restrict__ wihf,
                        const _Float16* __restrict__ whhf,
                        const float* __restrict__ bih,
                        const float* __restrict__ bhh,
                        float* __restrict__ out,
                        int N) {
  __shared__ _Float16 lds_ma[TN * DSZ];  // m_agg f16
  __shared__ _Float16 lds_s [TN * DSZ];  // S f16
  __shared__ float    lds_sf[TN * DSZ];  // S f32 (for z*S blend)
  __shared__ float    lds_cnt[TN];

  const int tid   = threadIdx.x;
  const int lane  = tid & 31;
  const int w     = tid >> 5;              // 0..7 -> column tile
  const int nbase = blockIdx.x * TN;

  if (tid < TN) {
    int nd = nbase + tid; if (nd >= N) nd = N - 1;
    lds_cnt[tid] = cnt[nd];
  }
#pragma unroll
  for (int i = 0; i < 16; ++i) {
    int q   = i * 256 + tid;               // 0..4095
    int row = q >> 7, col = q & 127;
    int nd = nbase + row; if (nd >= N) nd = N - 1;
    float sv = S[(size_t)nd * DSZ + col];
    float c  = cnt[nd];
    float ma = msum[(size_t)nd * DSZ + col] / fmaxf(c, 1.0f);
    lds_sf[q] = sv;
    lds_s [q] = (_Float16)sv;
    lds_ma[q] = (_Float16)ma;
  }
  __syncthreads();

  const int m    = lane & 15;
  const int cb   = (lane >= 16) ? 8 : 0;
  const int mrow = (lane >= 16) ? 8 : 0;

  v8f ax[3][MTG] = {};   // xr, xz, xn
  v8f ah[3][MTG] = {};   // hr, hz, hn
  for (int kt = 0; kt < KTG; ++kt) {
    v16h a_ma[MTG], a_s[MTG];
#pragma unroll
    for (int mt = 0; mt < MTG; ++mt) {
      a_ma[mt] = mk_a(&lds_ma[(mt * 16 + m) * DSZ + kt * 32 + cb]);
      a_s [mt] = mk_a(&lds_s [(mt * 16 + m) * DSZ + kt * 32 + cb]);
    }
#pragma unroll
    for (int g = 0; g < 3; ++g) {
      int nt = g * 8 + w;                 // 0..23
      v16h b_ih = ld_b(wihf, kt * NTG + nt, lane);
      v16h b_hh = ld_b(whhf, kt * NTG + nt, lane);
#pragma unroll
      for (int mt = 0; mt < MTG; ++mt) {
        ax[g][mt] = __builtin_amdgcn_wmma_f32_16x16x32_f16(
            false, a_ma[mt], false, b_ih, (short)0, ax[g][mt], false, false);
        ah[g][mt] = __builtin_amdgcn_wmma_f32_16x16x32_f16(
            false, a_s[mt],  false, b_hh, (short)0, ah[g][mt], false, false);
      }
    }
  }

  const int col = w * 16 + (lane & 15);    // 0..127
  const float bxr = bih[col],       bhr = bhh[col];
  const float bxz = bih[128 + col], bhz = bhh[128 + col];
  const float bxn = bih[256 + col], bhn = bhh[256 + col];
#pragma unroll
  for (int mt = 0; mt < MTG; ++mt) {
#pragma unroll
    for (int v = 0; v < 8; ++v) {
      int mm   = mt * 16 + v + mrow;
      int node = nbase + mm;
      float xr = ax[0][mt][v] + bxr;
      float xz = ax[1][mt][v] + bxz;
      float xn = ax[2][mt][v] + bxn;
      float hr = ah[0][mt][v] + bhr;
      float hz = ah[1][mt][v] + bhz;
      float hn = ah[2][mt][v] + bhn;
      float r  = 1.0f / (1.0f + __expf(-(xr + hr)));
      float z  = 1.0f / (1.0f + __expf(-(xz + hz)));
      float nn = tanhf(xn + r * hn);
      float so = lds_sf[mm * DSZ + col];
      float hnew = (1.0f - z) * nn + z * so;
      float res  = (lds_cnt[mm] > 0.0f) ? hnew : 1.0f;
      if (node < N) out[(size_t)node * DSZ + col] = res;
    }
  }
}

// ---- host launcher -------------------------------------------------------

extern "C" void kernel_launch(void* const* d_in, const int* in_sizes, int n_in,
                              void* d_out, int out_size, void* d_ws, size_t ws_size,
                              hipStream_t stream) {
  const float* S   = (const float*)d_in[0];
  const int*   src = (const int*)  d_in[1];
  const int*   dst = (const int*)  d_in[2];
  const float* dt  = (const float*)d_in[3];
  const float* ea  = (const float*)d_in[4];
  const float* W1  = (const float*)d_in[5];
  const float* b1  = (const float*)d_in[6];
  const float* W2  = (const float*)d_in[7];
  const float* b2  = (const float*)d_in[8];
  const float* Wih = (const float*)d_in[9];
  const float* Whh = (const float*)d_in[10];
  const float* bih = (const float*)d_in[11];
  const float* bhh = (const float*)d_in[12];

  const int N = in_sizes[0] / DSZ;
  const int E = in_sizes[1];

  // workspace layout (256B aligned chunks)
  char* ws = (char*)d_ws;
  size_t off = 0;
  auto take = [&](size_t bytes) {
    void* p = ws + off;
    off = (off + bytes + 255u) & ~(size_t)255u;
    return p;
  };
  float*    msum = (float*)   take((size_t)N * DSZ * sizeof(float));
  float*    cntb = (float*)   take((size_t)N * sizeof(float));
  _Float16* w1f  = (_Float16*)take((size_t)KT1 * NT1 * 512 * sizeof(_Float16));
  _Float16* w2f  = (_Float16*)take((size_t)KT2 * NT2 * 512 * sizeof(_Float16));
  _Float16* wihf = (_Float16*)take((size_t)KTG * NTG * 512 * sizeof(_Float16));
  _Float16* whhf = (_Float16*)take((size_t)KTG * NTG * 512 * sizeof(_Float16));

  // 1) zero msum (N*128) and cnt (N)
  tgn_zero<<<4096, 256, 0, stream>>>(msum, (size_t)N * DSZ);
  tgn_zero<<<256,  256, 0, stream>>>(cntb, (size_t)N);

  // 2) pack weights into WMMA B fragments (f16)
  tgn_pack<<<288, 256, 0, stream>>>(W1,  w1f,  KT1, NT1, 385, 256);
  tgn_pack<<<96,  256, 0, stream>>>(W2,  w2f,  KT2, NT2, 192, -1);
  tgn_pack<<<96,  256, 0, stream>>>(Wih, wihf, KTG, NTG, 128, -1);
  tgn_pack<<<96,  256, 0, stream>>>(Whh, whhf, KTG, NTG, 128, -1);

  // 3) edge MLP + mean-aggregation scatter
  int eblocks = (E + TE - 1) / TE;
  tgn_edge<<<eblocks, 128, 0, stream>>>(S, src, dst, dt, ea, W1, b1, b2,
                                        w1f, w2f, msum, cntb, E);

  // 4) GRU memory update
  int nblocks = (N + TN - 1) / TN;
  tgn_gru<<<nblocks, 256, 0, stream>>>(S, msum, cntb, wihf, whhf, bih, bhh,
                                       (float*)d_out, N);
}
